// MultiHeadAttention_1726576853596
// MI455X (gfx1250) — matrix operand order repaired and checked by reading, not compiled, not run on hardware
//
#include <hip/hip_runtime.h>

// ---------------------------------------------------------------------------
// Multi-head causal attention for gfx1250 (MI455X), bf16 WMMA pipeline with
// async global->LDS double-buffered V staging.
// B=2, S=4096, DM=768, H=12, DH=64.
// ---------------------------------------------------------------------------

#define BB     2
#define SS     4096
#define DMODEL 768
#define NH     12
#define DHEAD  64
#define NQKV   (3 * DMODEL)   // 2304

typedef __attribute__((ext_vector_type(16))) __bf16 v16bf;
typedef __attribute__((ext_vector_type(2)))  __bf16 v2bf;
typedef __attribute__((ext_vector_type(8)))  float  v8f;
typedef __attribute__((ext_vector_type(8)))  unsigned int v8u;
typedef unsigned short u16;
typedef unsigned int   u32;
typedef unsigned long long u64;

__device__ __forceinline__ u32 f2bf_bits(float f) {
  u32 u = __builtin_bit_cast(u32, f);
  return (u + 0x7FFFu + ((u >> 16) & 1u)) >> 16;    // round-to-nearest-even
}
__device__ __forceinline__ u32 pack2bf(float lo, float hi) {
#if __has_builtin(__builtin_amdgcn_cvt_pk_bf16_f32)
  v2bf p = __builtin_amdgcn_cvt_pk_bf16_f32(lo, hi); // single packed convert
  return __builtin_bit_cast(u32, p);
#else
  return f2bf_bits(lo) | (f2bf_bits(hi) << 16);
#endif
}
__device__ __forceinline__ u32 packbits(u32 lo, u32 hi) {
  return (lo & 0xFFFFu) | (hi << 16);
}

// ---------------------------------------------------------------------------
// Kernel 1: qkv = x @ w_qkv + b_qkv    (M=8192, N=2304, K=768)
// One 16x16 WMMA tile per wave; Q/K/V written bf16 in [B,H,S,DH] layout.
// ---------------------------------------------------------------------------
__global__ __launch_bounds__(256)
void qkv_proj_kernel(const float* __restrict__ x, const float* __restrict__ w,
                     const float* __restrict__ bias,
                     u16* __restrict__ Qb, u16* __restrict__ Kb, u16* __restrict__ Vb)
{
  const u32 lane = threadIdx.x & 31u;
  const u32 wave = threadIdx.x >> 5;
  const u32 t    = blockIdx.x * 8u + wave;
  const u32 NT   = NQKV / 16;          // 144 column tiles
  const u32 mt   = t / NT;
  const u32 nt   = t % NT;
  const u32 half = lane >> 4;
  const u32 lm   = lane & 15u;

  const u32 arow = mt * 16 + lm;       // A-fragment row owned by this lane
  const u32 col  = nt * 16 + lm;       // B/C column owned by this lane

  v8f acc = {};
  for (u32 k0 = 0; k0 < DMODEL; k0 += 32) {
    // A fragment (x, fp32 -> packed bf16).  k = half*16 + g*8 + i
    v8u au;
    #pragma unroll
    for (u32 g = 0; g < 2; ++g) {
      const float4* p = (const float4*)(x + (size_t)arow * DMODEL + k0 + half * 16 + g * 8);
      float4 f0 = p[0], f1 = p[1];
      au[g * 4 + 0] = pack2bf(f0.x, f0.y);
      au[g * 4 + 1] = pack2bf(f0.z, f0.w);
      au[g * 4 + 2] = pack2bf(f1.x, f1.y);
      au[g * 4 + 3] = pack2bf(f1.z, f1.w);
    }
    // B fragment: w[k][col], k = half*16 + e (coalesced across the 16 lanes)
    v8u bu;
    #pragma unroll
    for (u32 e2 = 0; e2 < 8; ++e2) {
      u32 k = k0 + half * 16 + e2 * 2;
      bu[e2] = pack2bf(w[(size_t)k * NQKV + col], w[(size_t)(k + 1) * NQKV + col]);
    }
    acc = __builtin_amdgcn_wmma_f32_16x16x32_bf16(
        false, __builtin_bit_cast(v16bf, au), false, __builtin_bit_cast(v16bf, bu),
        (short)0, acc, false, false);
  }

  const float bv = bias[col];
  const u32 which = (col >= 2u * DMODEL) ? 2u : (col >= DMODEL ? 1u : 0u);
  const u32 dmc   = col - which * DMODEL;
  const u32 h     = dmc >> 6;          // /DHEAD
  const u32 d     = dmc & 63u;
  u16* dst = (which == 0) ? Qb : (which == 1) ? Kb : Vb;

  #pragma unroll
  for (u32 r = 0; r < 8; ++r) {
    u32 rg = mt * 16 + half * 8 + r;   // global row (C layout: m = half*8 + r)
    u32 b  = rg >> 12;                 // /SS
    u32 s  = rg & 4095u;               // %SS
    dst[((size_t)(b * NH + h) * SS + s) * DHEAD + d] = (u16)f2bf_bits(acc[r] + bv);
  }
}

// ---------------------------------------------------------------------------
// Async staging: copy one 32x64 bf16 V block (4KB) global -> LDS.
// INST_OFFSET is added to BOTH the LDS and global address (ISA 08 §4.4), and
// our per-issue delta is +512B on both sides -> one base address pair plus
// eight offset: immediates moves the whole block.
// ---------------------------------------------------------------------------
#define STAGE_ASYNC_B128(OFF)                                            \
  asm volatile("global_load_async_to_lds_b128 %0, %1, off offset:" #OFF  \
               :: "v"(loff), "v"(ga) : "memory")

__device__ __forceinline__ void stage_v_async(const u16* __restrict__ gsrc,
                                              u32 lds_byte_base, u32 lane)
{
  const u32 row = lane >> 3;           // 0..3
  const u32 seg = lane & 7u;           // 0..7 (16B segments)
  u32 loff = lds_byte_base + (row * 64 + seg * 8) * 2;
  u64 ga   = (u64)(uintptr_t)(gsrc + (size_t)row * DHEAD + seg * 8);
  STAGE_ASYNC_B128(0);    STAGE_ASYNC_B128(512);
  STAGE_ASYNC_B128(1024); STAGE_ASYNC_B128(1536);
  STAGE_ASYNC_B128(2048); STAGE_ASYNC_B128(2560);
  STAGE_ASYNC_B128(3072); STAGE_ASYNC_B128(3584);
}

// ---------------------------------------------------------------------------
// Kernel 2: causal flash attention.  One wave = one 16-row query tile.
// Key blocks of 32; QK^T = 4 WMMAs, P.V = 4 WMMAs per block.
// V blocks double-buffered in LDS via async DMA, overlapped with compute.
// Only the diagonal key block needs the causal mask (wave-uniform branch).
// ---------------------------------------------------------------------------
__global__ __launch_bounds__(256)
void attention_kernel(const u16* __restrict__ Qb, const u16* __restrict__ Kb,
                      const u16* __restrict__ Vb, u16* __restrict__ Ob)
{
  __shared__ u16 sV[8][2][32 * 64];    // 64 KB: per-wave double-buffered V block
  __shared__ u16 sP[8][16 * 32];       // 8 KB : per-wave P staging (C->A relayout)

  const u32 lane = threadIdx.x & 31u;
  const u32 wave = threadIdx.x >> 5;
  const u32 t    = blockIdx.x * 8u + wave;
  const u32 qt   = t & 255u;           // % (SS/16), wave-uniform
  const u32 bh   = t >> 8;             // flat (b,h)
  const u32 half = lane >> 4;
  const u32 lm   = lane & 15u;
  const size_t base = (size_t)bh * SS * DHEAD;

  // Q A-fragments for both K=32 chunks of DH=64 (16B vector loads)
  v16bf aq[2];
  #pragma unroll
  for (u32 c = 0; c < 2; ++c) {
    v8u qu;
    #pragma unroll
    for (u32 g = 0; g < 2; ++g) {
      const uint4* p = (const uint4*)(Qb + base + (size_t)(qt * 16 + lm) * DHEAD +
                                      c * 32 + half * 16 + g * 8);
      uint4 v = *p;
      qu[g * 4 + 0] = v.x; qu[g * 4 + 1] = v.y; qu[g * 4 + 2] = v.z; qu[g * 4 + 3] = v.w;
    }
    aq[c] = __builtin_bit_cast(v16bf, qu);
  }

  float mrow[8], lrow[8];
  v8f   accO[4] = {};
  #pragma unroll
  for (u32 r = 0; r < 8; ++r) { mrow[r] = -1.0e30f; lrow[r] = 0.0f; }

  const u32 nblocks = qt / 2 + 1;      // causal: keys <= qt*16+15
  const u32 ldsV0 = (u32)(uintptr_t)&sV[wave][0][0];   // LDS aperture: low 32 bits = offset
  const u32 ldsV1 = (u32)(uintptr_t)&sV[wave][1][0];

  stage_v_async(Vb + base, ldsV0, lane);               // prefetch first V block

  for (u32 jb = 0; jb < nblocks; ++jb) {
    const u32 buf = jb & 1u;

    // ---- scores S = Q @ K^T (16x32): overlaps with V DMA in flight ----
    v8f sc[2] = {};
    #pragma unroll
    for (u32 nh = 0; nh < 2; ++nh) {
      #pragma unroll
      for (u32 c = 0; c < 2; ++c) {
        const u16* kp = Kb + base + (size_t)(jb * 32 + nh * 16 + lm) * DHEAD +
                        c * 32 + half * 16;
        v16bf kf = *(const v16bf*)kp;  // contiguous along DH, 32B aligned
        sc[nh] = __builtin_amdgcn_wmma_f32_16x16x32_bf16(false, aq[c], false, kf,
                                                         (short)0, sc[nh], false, false);
      }
    }

    // ---- online softmax (C layout: row = half*8+r, col = lm) ----
    float pv0[8], pv1[8], tm[8];
    #pragma unroll
    for (u32 r = 0; r < 8; ++r) {
      pv0[r] = sc[0][r] * 0.125f;      // 1/sqrt(DH)
      pv1[r] = sc[1][r] * 0.125f;
    }
    if (jb + 1 == nblocks) {           // diagonal block only (wave-uniform)
      #pragma unroll
      for (u32 r = 0; r < 8; ++r) {
        int qrow = (int)(qt * 16 + half * 8 + r);
        if ((int)(jb * 32 + lm)      > qrow) pv0[r] = -1.0e30f;
        if ((int)(jb * 32 + 16 + lm) > qrow) pv1[r] = -1.0e30f;
      }
    }
    #pragma unroll
    for (u32 r = 0; r < 8; ++r) {
      tm[r] = fmaxf(pv0[r], pv1[r]);
      #pragma unroll
      for (int off = 1; off < 16; off <<= 1)
        tm[r] = fmaxf(tm[r], __shfl_xor(tm[r], off, 16));
    }
    #pragma unroll
    for (u32 r = 0; r < 8; ++r) {
      float nm    = fmaxf(mrow[r], tm[r]);
      float alpha = __expf(mrow[r] - nm);
      float p0    = __expf(pv0[r] - nm);
      float p1    = __expf(pv1[r] - nm);
      mrow[r] = nm;
      u32 m = half * 8 + r;
      sP[wave][m * 32 + lm]      = (u16)f2bf_bits(p0);
      sP[wave][m * 32 + 16 + lm] = (u16)f2bf_bits(p1);
      float sum = p0 + p1;
      #pragma unroll
      for (int off = 1; off < 16; off <<= 1) sum += __shfl_xor(sum, off, 16);
      lrow[r] = lrow[r] * alpha + sum;
      #pragma unroll
      for (u32 nt = 0; nt < 4; ++nt) accO[nt][r] *= alpha;
    }
    asm volatile("" ::: "memory");     // order sP stores before sP loads (DS is in-order)

    // ---- issue next V DMA, then wait for the current buffer ----
    if (jb + 1 < nblocks) {
      asm volatile("s_wait_dscnt 0x0" ::: "memory");   // close WAR on buf^1
      stage_v_async(Vb + base + (size_t)(jb + 1) * 32 * DHEAD,
                    (buf ? ldsV0 : ldsV1), lane);
      asm volatile("s_wait_asynccnt 0x8" ::: "memory"); // oldest 8 (current buf) done
    } else {
      asm volatile("s_wait_asynccnt 0x0" ::: "memory");
    }

    // ---- P A-fragment from LDS (A layout: row = lm, k = half*16 + g*8 + i) ----
    v8u pu;
    #pragma unroll
    for (u32 g = 0; g < 2; ++g) {
      const uint4* pp = (const uint4*)&sP[wave][lm * 32 + half * 16 + g * 8];
      uint4 v = *pp;
      pu[g * 4 + 0] = v.x; pu[g * 4 + 1] = v.y; pu[g * 4 + 2] = v.z; pu[g * 4 + 3] = v.w;
    }
    v16bf ap = __builtin_bit_cast(v16bf, pu);

    // ---- O += P @ V from LDS (4 column tiles of 16) ----
    const u16* vs = &sV[wave][buf][0];
    #pragma unroll
    for (u32 nt = 0; nt < 4; ++nt) {
      v8u vu;
      #pragma unroll
      for (u32 e2 = 0; e2 < 8; ++e2) {
        u32 k = half * 16 + e2 * 2;
        vu[e2] = packbits(vs[k * 64 + nt * 16 + lm], vs[(k + 1) * 64 + nt * 16 + lm]);
      }
      accO[nt] = __builtin_amdgcn_wmma_f32_16x16x32_bf16(
          false, ap, false, __builtin_bit_cast(v16bf, vu),
          (short)0, accO[nt], false, false);
    }
  }

  // ---- epilogue: normalize, store bf16 in [B,S,H,DH] (== [B,S,DM] flat) ----
  const u32 b = (bh >= NH) ? 1u : 0u;
  const u32 h = bh - b * NH;
  u16* obase = Ob + ((size_t)(b * SS + qt * 16 + half * 8) * NH + h) * DHEAD + lm;
  #pragma unroll
  for (u32 nt = 0; nt < 4; ++nt) {
    #pragma unroll
    for (u32 r = 0; r < 8; ++r) {
      float o = accO[nt][r] / lrow[r];
      obase[(size_t)r * (NH * DHEAD) + nt * 16] = (u16)f2bf_bits(o);
    }
  }
}

// ---------------------------------------------------------------------------
// Kernel 3: out = attn @ w_out + b_out   (M=8192, N=768, K=768), fp32 output
// ---------------------------------------------------------------------------
__global__ __launch_bounds__(256)
void out_proj_kernel(const u16* __restrict__ Ab, const float* __restrict__ w,
                     const float* __restrict__ bias, float* __restrict__ out)
{
  const u32 lane = threadIdx.x & 31u;
  const u32 wave = threadIdx.x >> 5;
  const u32 t    = blockIdx.x * 8u + wave;
  const u32 NT   = DMODEL / 16;        // 48
  const u32 mt   = t / NT;
  const u32 nt   = t % NT;
  const u32 half = lane >> 4;
  const u32 lm   = lane & 15u;

  const u32 arow = mt * 16 + lm;
  const u32 col  = nt * 16 + lm;

  v8f acc = {};
  for (u32 k0 = 0; k0 < DMODEL; k0 += 32) {
    v8u au;
    #pragma unroll
    for (u32 g = 0; g < 2; ++g) {
      const uint4* p = (const uint4*)(Ab + (size_t)arow * DMODEL + k0 + half * 16 + g * 8);
      uint4 v = *p;
      au[g * 4 + 0] = v.x; au[g * 4 + 1] = v.y; au[g * 4 + 2] = v.z; au[g * 4 + 3] = v.w;
    }
    v8u bu;
    #pragma unroll
    for (u32 e2 = 0; e2 < 8; ++e2) {
      u32 k = k0 + half * 16 + e2 * 2;
      bu[e2] = pack2bf(w[(size_t)k * DMODEL + col], w[(size_t)(k + 1) * DMODEL + col]);
    }
    acc = __builtin_amdgcn_wmma_f32_16x16x32_bf16(
        false, __builtin_bit_cast(v16bf, au), false, __builtin_bit_cast(v16bf, bu),
        (short)0, acc, false, false);
  }

  const float bv = bias[col];
  #pragma unroll
  for (u32 r = 0; r < 8; ++r) {
    u32 rg = mt * 16 + half * 8 + r;
    out[(size_t)rg * DMODEL + col] = acc[r] + bv;
  }
}

// ---------------------------------------------------------------------------
// Launch
// ---------------------------------------------------------------------------
extern "C" void kernel_launch(void* const* d_in, const int* in_sizes, int n_in,
                              void* d_out, int out_size, void* d_ws, size_t ws_size,
                              hipStream_t stream)
{
  (void)in_sizes; (void)n_in; (void)out_size; (void)ws_size;

  const float* x     = (const float*)d_in[0];
  const float* w_qkv = (const float*)d_in[1];
  const float* b_qkv = (const float*)d_in[2];
  const float* w_out = (const float*)d_in[3];
  const float* b_out = (const float*)d_in[4];
  float*       out   = (float*)d_out;

  const size_t qkvElems = (size_t)BB * NH * SS * DHEAD;   // 6,291,456
  u16* Qb = (u16*)d_ws;
  u16* Kb = Qb + qkvElems;
  u16* Vb = Kb + qkvElems;
  u16* Ob = Vb + qkvElems;                                // total ~50.3 MB bf16

  const int M = BB * SS;                                  // 8192
  qkv_proj_kernel<<<(M / 16) * (NQKV / 16) / 8, 256, 0, stream>>>(x, w_qkv, b_qkv, Qb, Kb, Vb);
  attention_kernel<<<BB * NH * (SS / 16) / 8, 256, 0, stream>>>(Qb, Kb, Vb, Ob);
  out_proj_kernel<<<(M / 16) * (DMODEL / 16) / 8, 256, 0, stream>>>(Ob, w_out, b_out, out);
}
